// myxLSTM_18356690223644
// MI455X (gfx1250) — matrix operand order repaired and checked by reading, not compiled, not run on hardware
//
#include <hip/hip_runtime.h>
#include <math.h>

// ---------------- model dimensions ----------------
#define BBATCH 16
#define L0 5600
#define L1 5599
#define P1 2799
#define L2A 2797
#define P2 1398
#define L3A 1396
#define SL  698
#define SPAD 704          // S padded to multiple of 16 for WMMA tiles
#define STILES 44
#define DD  128
#define BS  (BBATCH*SL)   // 11168 rows, multiple of 16
#define NHH 4
#define DHH 32
#define MI  256
#define MDH 64
#define FFD 192

typedef __attribute__((ext_vector_type(16))) __bf16 v16bf;
typedef __attribute__((ext_vector_type(8)))  float  v8f;

// ---------------- device helpers ----------------
__device__ __forceinline__ float leakyf(float x){ return x >= 0.f ? x : 0.01f*x; }
__device__ __forceinline__ float sigmoidf_(float x){ return 1.f/(1.f+expf(-x)); }
__device__ __forceinline__ float siluf(float x){ return x * sigmoidf_(x); }
__device__ __forceinline__ float logsigf(float x){
    // numerically stable log(sigmoid(x))
    return (x <= 0.f) ? (x - log1pf(expf(x))) : (-log1pf(expf(-x)));
}
__device__ __forceinline__ float geluf(float x){
    return 0.5f*x*(1.f + erff(x*0.70710678118654752f));
}
__device__ __forceinline__ unsigned short f2bf(float f){
    unsigned int u = __float_as_uint(f);
    unsigned int lsb = (u >> 16) & 1u;
    u += 0x7fffu + lsb;                 // round to nearest even
    return (unsigned short)(u >> 16);
}

// ================= conv front-end =================
__global__ void conv1_kernel(const float* __restrict__ x, const float* __restrict__ wa,
                             const float* __restrict__ wb, float* __restrict__ out, int total){
    int idx = blockIdx.x*blockDim.x + threadIdx.x;
    if (idx >= total) return;
    int j = idx % L1; int o = (idx / L1) % 32; int b = idx / (L1*32);
    float a = 0.f;
    if (o < 16) {                       // conv1a: k=4, pad=1
        #pragma unroll
        for (int i = 0; i < 2; ++i)
            #pragma unroll
            for (int k = 0; k < 4; ++k) {
                int p = j + k - 1;
                if (p >= 0 && p < L0) a += x[((size_t)b*L0 + p)*2 + i] * wa[(o*2+i)*4 + k];
            }
    } else {                            // conv1b: k=2, pad=0
        int oo = o - 16;
        #pragma unroll
        for (int i = 0; i < 2; ++i)
            #pragma unroll
            for (int k = 0; k < 2; ++k)
                a += x[((size_t)b*L0 + (j+k))*2 + i] * wb[(oo*2+i)*2 + k];
    }
    out[idx] = a;
}

__global__ void bn_stats_kernel(const float* __restrict__ in, float* __restrict__ stats,
                                int C, int L){
    int c = blockIdx.x; int tid = threadIdx.x;
    int N = BBATCH * L;
    float s = 0.f, s2 = 0.f;
    for (int i = tid; i < N; i += 256) {
        int b = i / L, l = i % L;
        float v = in[((size_t)b*C + c)*L + l];
        s += v; s2 += v*v;
    }
    __shared__ float rs[256], rq[256];
    rs[tid] = s; rq[tid] = s2; __syncthreads();
    for (int off = 128; off >= 1; off >>= 1) {
        if (tid < off) { rs[tid] += rs[tid+off]; rq[tid] += rq[tid+off]; }
        __syncthreads();
    }
    if (tid == 0) {
        float mu = rs[0] / (float)N;
        stats[c] = mu;
        stats[C + c] = rq[0] / (float)N - mu*mu;
    }
}

__global__ void bn_pool_kernel(const float* __restrict__ in, const float* __restrict__ g,
                               const float* __restrict__ bb, const float* __restrict__ stats,
                               float* __restrict__ out, int C, int Lin, int Lo, int transpose, int total){
    int idx = blockIdx.x*blockDim.x + threadIdx.x;
    if (idx >= total) return;
    int j = idx % Lo; int c = (idx / Lo) % C; int b = idx / (Lo*C);
    float mu = stats[c], var = stats[C + c];
    float sc = rsqrtf(var + 1e-5f) * g[c];
    float bo = bb[c] - mu*sc;
    const float* ip = in + ((size_t)b*C + c)*Lin + 2*j;
    float v = fmaxf(leakyf(ip[0]*sc + bo), leakyf(ip[1]*sc + bo));
    if (transpose) out[((size_t)b*Lo + j)*C + c] = v;
    else           out[((size_t)b*C + c)*Lo + j] = v;
}

__global__ void conv3_kernel(const float* __restrict__ in, const float* __restrict__ w,
                             float* __restrict__ out, int Cin, int Cout, int Lin, int total){
    int idx = blockIdx.x*blockDim.x + threadIdx.x;
    if (idx >= total) return;
    int Lout = Lin - 2;
    int j = idx % Lout; int o = (idx / Lout) % Cout; int b = idx / (Lout*Cout);
    const float* ip = in + (size_t)b*Cin*Lin + j;
    const float* wp = w + (size_t)o*Cin*3;
    float a = 0.f;
    for (int i = 0; i < Cin; ++i)
        a += ip[(size_t)i*Lin]*wp[i*3] + ip[(size_t)i*Lin+1]*wp[i*3+1] + ip[(size_t)i*Lin+2]*wp[i*3+2];
    out[idx] = a;
}

// ================= common pieces =================
__global__ void layernorm128(const float* __restrict__ in, const float* __restrict__ w,
                             float* __restrict__ out, float eps){
    int row = blockIdx.x, tid = threadIdx.x;      // blockDim = 128
    float v = in[(size_t)row*DD + tid];
    __shared__ float red[128];
    red[tid] = v; __syncthreads();
    for (int off = 64; off >= 1; off >>= 1) { if (tid < off) red[tid] += red[tid+off]; __syncthreads(); }
    float mu = red[0] * (1.f/128.f); __syncthreads();
    float dv = v - mu;
    red[tid] = dv*dv; __syncthreads();
    for (int off = 64; off >= 1; off >>= 1) { if (tid < off) red[tid] += red[tid+off]; __syncthreads(); }
    float var = red[0] * (1.f/128.f);
    out[(size_t)row*DD + tid] = dv * rsqrtf(var + eps) * w[tid];
}

// causal depthwise conv (k=3) + bias + SiLU; input row stride ldin, output compact C
__global__ void causal_dw_silu(const float* __restrict__ in, int ldin, int C,
                               const float* __restrict__ w, const float* __restrict__ bias,
                               float* __restrict__ out, int total){
    int idx = blockIdx.x*blockDim.x + threadIdx.x;
    if (idx >= total) return;
    int c = idx % C; int rs = idx / C; int s = rs % SL; int b = rs / SL;
    float a = bias[c];
    #pragma unroll
    for (int k = 0; k < 3; ++k) {
        int sp = s - 2 + k;
        if (sp >= 0) a += in[((size_t)(b*SL + sp))*ldin + c] * w[c*3 + k];
    }
    out[(size_t)rs*C + c] = siluf(a);
}

__global__ void f32_to_bf16_kernel(const float* __restrict__ in, unsigned short* __restrict__ out, int total){
    int idx = blockIdx.x*blockDim.x + threadIdx.x;
    if (idx < total) out[idx] = f2bf(in[idx]);
}

// W (K,N) f32 -> Wt (N,K) bf16
__global__ void transpose_to_bf16(const float* __restrict__ W, unsigned short* __restrict__ Wt,
                                  int K, int N){
    int idx = blockIdx.x*blockDim.x + threadIdx.x;
    if (idx >= K*N) return;
    int k = idx / N, n = idx % N;
    Wt[(size_t)n*K + k] = f2bf(W[idx]);
}

// ================= WMMA GEMM: C(M,N) = A(M,K) * Bt(N,K)^T, bf16 in, f32 acc ===========
__global__ void wmma_gemm_abT(const unsigned short* __restrict__ A, const unsigned short* __restrict__ Bt,
                              float* __restrict__ C, int M, int N, int K, int accumulate){
    int wave = blockIdx.x * (blockDim.x >> 5) + (threadIdx.x >> 5);
    int lane = threadIdx.x & 31;
    int tilesN = N >> 4;
    int tiles  = (M >> 4) * tilesN;
    if (wave >= tiles) return;
    int tm = wave / tilesN, tn = wave % tilesN;
    int rsel = lane >> 4;          // half-wave select
    int r15  = lane & 15;
    const unsigned short* arow = A  + (size_t)(tm*16 + r15) * K;
    const unsigned short* brow = Bt + (size_t)(tn*16 + r15) * K;
    v8f acc = {};
    for (int k0 = 0; k0 < K; k0 += 32) {
        // prefetch the next K-block of the streamed operands (global_prefetch_b8)
        __builtin_prefetch(arow + k0 + 32, 0, 0);
        __builtin_prefetch(brow + k0 + 32, 0, 0);
        union { v16bf v; uint4 u[2]; } af, bf;
        af.u[0] = *(const uint4*)(arow + k0 + 8*rsel);        // K = k0+8*sel .. +7
        af.u[1] = *(const uint4*)(arow + k0 + 16 + 8*rsel);   // K = k0+16+8*sel .. +7
        bf.u[0] = *(const uint4*)(brow + k0 + 8*rsel);        // K = k0+8*sel .. +7
        bf.u[1] = *(const uint4*)(brow + k0 + 16 + 8*rsel);   // K = k0+16+8*sel .. +7
        acc = __builtin_amdgcn_wmma_f32_16x16x32_bf16(false, af.v, false, bf.v,
                                                      (short)0, acc, false, false);
    }
    float* crow = C + (size_t)(tm*16 + 8*rsel) * N + tn*16 + r15;
    #pragma unroll
    for (int r = 0; r < 8; ++r) {
        float v = acc[r];
        if (accumulate) v += crow[(size_t)r * N];
        crow[(size_t)r * N] = v;
    }
}

// ================= sLSTM =================
__global__ void slstm_gates_kernel(const float* __restrict__ xi, const float* __restrict__ xcv,
                                   const float* __restrict__ Wi, const float* __restrict__ Wf,
                                   const float* __restrict__ Wz, const float* __restrict__ Wo,
                                   float* __restrict__ gates, int total){
    int idx = blockIdx.x*blockDim.x + threadIdx.x;
    if (idx >= total) return;
    int e = idx & 31; int g = (idx >> 5) & 3; int h = (idx >> 7) & 3; int rs = idx >> 9;
    const float* x = ((g < 2) ? xcv : xi) + (size_t)rs*DD + h*DHH;
    const float* W = (g == 0 ? Wi : g == 1 ? Wf : g == 2 ? Wz : Wo) + ((size_t)h*DHH + e)*DHH;
    float a = 0.f;
    #pragma unroll 8
    for (int d = 0; d < DHH; ++d) a += x[d]*W[d];
    gates[idx] = a;
}

__global__ void slstm_scan_kernel(const float* __restrict__ gates, const float* __restrict__ R,
                                  const float* __restrict__ bias, float* __restrict__ hs){
    int bh = blockIdx.x; int h = bh & 3; int b = bh >> 2;
    int tid = threadIdx.x;              // 128 threads: tid = g*32+e
    int g = tid >> 5, e = tid & 31;
    __shared__ float hprev[32], pre[128];
    const float* Rp = R + (size_t)h*DHH*4*DHH + g*DHH + e;   // stride 128 over d
    float bval = bias[(h*4 + g)*DHH + e];
    float c = 0.f, n = 0.f, m = 0.f;
    if (tid < 32) hprev[tid] = 0.f;
    __syncthreads();
    for (int s = 0; s < SL; ++s) {
        float a = gates[(((size_t)(b*SL + s)*4 + h)*4 + g)*DHH + e] + bval;
        #pragma unroll 8
        for (int d = 0; d < DHH; ++d) a += hprev[d] * Rp[(size_t)d*128];
        pre[tid] = a;
        __syncthreads();
        if (tid < 32) {
            float ir = pre[tid], fr = pre[32+tid], zr = pre[64+tid], og = pre[96+tid];
            float lfm = m + logsigf(fr);
            float mnew = fmaxf(ir, lfm);
            float ig = expf(ir - mnew), fg = expf(lfm - mnew);
            float cnew = fg*c + ig*tanhf(zr);
            float nnew = fg*n + ig;
            float hnew = sigmoidf_(og) * cnew / nnew;
            c = cnew; n = nnew; m = mnew;
            hprev[tid] = hnew;
            hs[((size_t)(b*SL + s)*4 + h)*DHH + tid] = hnew;
        }
        __syncthreads();
    }
}

__global__ void slstm_headnorm_add(const float* __restrict__ hs, const float* __restrict__ gnw,
                                   float* __restrict__ seq){
    int row = blockIdx.x; int tid = threadIdx.x;  // 128 threads, wave == head (wave32)
    float v = hs[(size_t)row*DD + tid];
    float s = v;
    for (int off = 16; off >= 1; off >>= 1) s += __shfl_xor(s, off, 32);
    float mu = s * (1.f/32.f);
    float dv = v - mu;
    float q = dv*dv;
    for (int off = 16; off >= 1; off >>= 1) q += __shfl_xor(q, off, 32);
    float var = q * (1.f/32.f);
    seq[(size_t)row*DD + tid] += dv * rsqrtf(var + 1e-6f) * gnw[tid];
}

// ================= FFN =================
__global__ void ff_act_kernel(const float* __restrict__ gup, float* __restrict__ mid, int total){
    int idx = blockIdx.x*blockDim.x + threadIdx.x;
    if (idx >= total) return;
    int j = idx % FFD; int r = idx / FFD;
    float g = gup[(size_t)r*384 + j], u = gup[(size_t)r*384 + 192 + j];
    mid[idx] = geluf(g) * u;
}

// ================= mLSTM =================
__global__ void bdiag_kernel(const float* __restrict__ t_in, int ldin,
                             const float* __restrict__ W, float* __restrict__ out, int total){
    int idx = blockIdx.x*blockDim.x + threadIdx.x;
    if (idx >= total) return;
    int c = idx % MI; int r = idx / MI;
    int n = c >> 2, o = c & 3;
    const float* x = t_in + (size_t)r*ldin + n*4;
    const float* w = W + ((size_t)n*4 + o)*4;
    out[idx] = x[0]*w[0] + x[1]*w[1] + x[2]*w[2] + x[3]*w[3];
}

__global__ void mlstm_ifpre(const float* __restrict__ q, const float* __restrict__ k,
                            const float* __restrict__ v, const float* __restrict__ Wi,
                            const float* __restrict__ bi, const float* __restrict__ Wf,
                            const float* __restrict__ bfv, float* __restrict__ i_pre,
                            float* __restrict__ f_pre, int total){
    int idx = blockIdx.x*blockDim.x + threadIdx.x;
    if (idx >= total) return;                       // (b,h,s)
    int s = idx % SL; int h = (idx / SL) % NHH; int b = idx / (SL*NHH);
    size_t row = (size_t)(b*SL + s)*MI;
    const float* wi = Wi + (size_t)h*3*MI;
    const float* wf = Wf + (size_t)h*3*MI;
    float si = bi[h], sf = bfv[h];
    for (int j = 0; j < MI; ++j) { float t = q[row+j]; si += t*wi[j];        sf += t*wf[j]; }
    for (int j = 0; j < MI; ++j) { float t = k[row+j]; si += t*wi[MI+j];     sf += t*wf[MI+j]; }
    for (int j = 0; j < MI; ++j) { float t = v[row+j]; si += t*wi[2*MI+j];   sf += t*wf[2*MI+j]; }
    i_pre[idx] = si; f_pre[idx] = sf;
}

// cumulative logsigmoid + analytic running row-max of log_D
__global__ void mlstm_scanstats(const float* __restrict__ f_pre, const float* __restrict__ i_pre,
                                float* __restrict__ lfc, float* __restrict__ mx){
    int bh = blockIdx.x*blockDim.x + threadIdx.x;
    if (bh >= BBATCH*NHH) return;
    size_t base = (size_t)bh*SL;
    float cum = 0.f, cmax = -3.4e38f;
    for (int s = 0; s < SL; ++s) {
        cum += logsigf(f_pre[base+s]);
        lfc[base+s] = cum;
        float a = i_pre[base+s] - cum;
        cmax = fmaxf(cmax, a);
        mx[base+s] = cum + cmax;        // max_log_D[s]
    }
}

__global__ void pack_heads_bf16(const float* __restrict__ src, unsigned short* __restrict__ dst,
                                float scale, int total){
    int idx = blockIdx.x*blockDim.x + threadIdx.x;
    if (idx >= total) return;
    int d = idx & 63; int sp = (idx >> 6) % SPAD; int bh = idx / (SPAD*64);
    int h = bh & 3, b = bh >> 2;
    float v = 0.f;
    if (sp < SL) v = src[((size_t)(b*SL + sp))*MI + h*MDH + d] * scale;
    dst[idx] = f2bf(v);
}

__global__ void pack_heads_f32(const float* __restrict__ src, float* __restrict__ dst, int total){
    int idx = blockIdx.x*blockDim.x + threadIdx.x;
    if (idx >= total) return;
    int d = idx & 63; int sp = (idx >> 6) % SPAD; int bh = idx / (SPAD*64);
    int h = bh & 3, b = bh >> 2;
    dst[idx] = (sp < SL) ? src[((size_t)(b*SL + sp))*MI + h*MDH + d] : 0.f;
}

// one block per (b,h,s-tile); 8 waves; WMMA q*k^T, decay weights in LDS, normalizer,
// then async-LDS staged V multiply (global_load_async_to_lds_b128 + s_wait_asynccnt)
__global__ void mlstm_attn(const unsigned short* __restrict__ qhb, const unsigned short* __restrict__ khb,
                           const float* __restrict__ vhp, const float* __restrict__ i_pre,
                           const float* __restrict__ lfc, const float* __restrict__ mx,
                           float* __restrict__ ht){
    int blk = blockIdx.x;
    int st = blk % STILES;
    int bh = blk / STILES;
    int h = bh & 3, b = bh >> 2;
    extern __shared__ float smem[];
    float* Cs = smem;                 // [16][SPAD]
    float* rowinv = smem + 16*SPAD;   // [16]
    float* Vs = rowinv + 16;          // [8][64] async-staged V chunk
    int tid = threadIdx.x;
    int wave = tid >> 5, lane = tid & 31;
    int rsel = lane >> 4, r15 = lane & 15;

    const unsigned short* qbase = qhb + (size_t)bh * SPAD * MDH;
    const unsigned short* kbase = khb + (size_t)bh * SPAD * MDH;
    const float* ip = i_pre + (size_t)bh * SL;
    const float* lf = lfc   + (size_t)bh * SL;
    const float* mxp = mx   + (size_t)bh * SL;

    int ntt = st + 1;
    const unsigned short* arow = qbase + (size_t)(st*16 + r15) * MDH;
    union { v16bf v; uint4 u[2]; } af0, af1;
    af0.u[0] = *(const uint4*)(arow + 0  + 8*rsel);
    af0.u[1] = *(const uint4*)(arow + 16 + 8*rsel);
    af1.u[0] = *(const uint4*)(arow + 32 + 8*rsel);
    af1.u[1] = *(const uint4*)(arow + 48 + 8*rsel);

    for (int tt = wave; tt < ntt; tt += 8) {
        const unsigned short* brow = kbase + (size_t)(tt*16 + r15) * MDH;
        union { v16bf v; uint4 u[2]; } bf0, bf1;
        bf0.u[0] = *(const uint4*)(brow + 0  + 8*rsel);
        bf0.u[1] = *(const uint4*)(brow + 16 + 8*rsel);
        bf1.u[0] = *(const uint4*)(brow + 32 + 8*rsel);
        bf1.u[1] = *(const uint4*)(brow + 48 + 8*rsel);
        v8f acc = {};
        acc = __builtin_amdgcn_wmma_f32_16x16x32_bf16(false, af0.v, false, bf0.v, (short)0, acc, false, false);
        acc = __builtin_amdgcn_wmma_f32_16x16x32_bf16(false, af1.v, false, bf1.v, (short)0, acc, false, false);
        int t = tt*16 + r15;
        #pragma unroll
        for (int r = 0; r < 8; ++r) {
            int m = r + 8*rsel;
            int s = st*16 + m;
            float val = 0.f;
            if (s < SL && t < SL && t <= s) {
                float logD = lf[s] - lf[t] + ip[t] - mxp[s];
                val = acc[r] * expf(logD);     // 1/sqrt(64) folded into q pack
            }
            Cs[m*SPAD + t] = val;
        }
    }
    __syncthreads();
    int limit = ntt * 16;
    for (int m = wave; m < 16; m += 8) {
        float ssum = 0.f;
        for (int t = lane; t < limit; t += 32) ssum += Cs[m*SPAD + t];
        for (int off = 16; off >= 1; off >>= 1) ssum += __shfl_xor(ssum, off, 32);
        if (lane == 0) {
            int s = st*16 + m;
            float norm = 1.f;
            if (s < SL) norm = fmaxf(fabsf(ssum), expf(-mxp[s])) + 1e-6f;
            rowinv[m] = 1.f / norm;
        }
    }
    __syncthreads();
    int d = tid & 63;
    int m0 = tid >> 6;                       // rows m0, m0+4, m0+8, m0+12
    unsigned vs_base = (unsigned)(size_t)Vs; // low 32 bits of flat shared ptr = LDS byte offset
    const float* vbase = vhp + (size_t)bh * SPAD * MDH;
    float a0 = 0.f, a1 = 0.f, a2 = 0.f, a3 = 0.f;
    for (int tc = 0; tc < limit; tc += 8) {  // limit is a multiple of 16
        if (tid < 128) {                     // 128 lanes x 16B = 8 rows x 64 floats
            unsigned long long gsrc = (unsigned long long)(vbase + (size_t)tc*MDH + tid*4);
            unsigned ldsoff = vs_base + tid*16u;
            asm volatile("global_load_async_to_lds_b128 %0, %1, off"
                         :: "v"(ldsoff), "v"(gsrc) : "memory");
        }
        asm volatile("s_wait_asynccnt 0x0" ::: "memory");
        __syncthreads();
        #pragma unroll
        for (int tt2 = 0; tt2 < 8; ++tt2) {
            int t = tc + tt2;
            float vv = Vs[tt2*MDH + d];
            a0 += Cs[(m0     )*SPAD + t] * vv;
            a1 += Cs[(m0 +  4)*SPAD + t] * vv;
            a2 += Cs[(m0 +  8)*SPAD + t] * vv;
            a3 += Cs[(m0 + 12)*SPAD + t] * vv;
        }
        __syncthreads();
    }
    float av[4] = {a0, a1, a2, a3};
    #pragma unroll
    for (int i = 0; i < 4; ++i) {
        int m = m0 + 4*i;
        int s = st*16 + m;
        if (s < SL) ht[((size_t)(b*SL + s))*MI + h*MDH + d] = av[i] * rowinv[m];
    }
}

// headnorm(ht) * w + skip*xcv2, * silu(zg) -> bf16 for down-projection GEMM
__global__ void mlstm_post(const float* __restrict__ ht, const float* __restrict__ normw,
                           const float* __restrict__ skip, const float* __restrict__ xcv2,
                           const float* __restrict__ up2, unsigned short* __restrict__ outbf){
    int row = blockIdx.x; int tid = threadIdx.x;   // 256 threads: tid = h*64+d
    int h = tid >> 6, d = tid & 63;
    float v = ht[(size_t)row*MI + tid];
    __shared__ float red[256];
    red[tid] = v; __syncthreads();
    for (int off = 32; off >= 1; off >>= 1) { if (d < off) red[tid] += red[tid+off]; __syncthreads(); }
    float mu = red[h*64] * (1.f/64.f);
    __syncthreads();
    float dv = v - mu;
    red[tid] = dv*dv; __syncthreads();
    for (int off = 32; off >= 1; off >>= 1) { if (d < off) red[tid] += red[tid+off]; __syncthreads(); }
    float var = red[h*64] * (1.f/64.f);
    float xn = dv * rsqrtf(var + 1e-6f);
    float hn = xn * normw[tid] + skip[tid] * xcv2[(size_t)row*MI + tid];
    float z = up2[(size_t)row*512 + MI + tid];
    outbf[(size_t)row*MI + tid] = f2bf(hn * siluf(z));
}

// ================= classifier head =================
__global__ void head_kernel(const float* __restrict__ xi, const float* __restrict__ fc1w,
                            const float* __restrict__ fc1b, const float* __restrict__ fc2w,
                            const float* __restrict__ fc2b, float* __restrict__ out){
    int b = blockIdx.x; int tid = threadIdx.x;     // 64 threads
    __shared__ float e[128], t1[64];
    const float* src = xi + ((size_t)(b*SL + SL - 1))*DD;
    e[tid] = src[tid]; e[tid+64] = src[tid+64];
    __syncthreads();
    float a = fc1b[tid];
    for (int c = 0; c < 128; ++c) a += e[c]*fc1w[c*64 + tid];
    t1[tid] = leakyf(a);
    __syncthreads();
    if (tid < 5) {
        float o = fc2b[tid];
        for (int k = 0; k < 64; ++k) o += t1[k]*fc2w[k*5 + tid];
        out[b*5 + tid] = o;
    }
}

// ================= host orchestration =================
static inline int grid1d(long long total){ return (int)((total + 255) / 256); }

extern "C" void kernel_launch(void* const* d_in, const int* in_sizes, int n_in,
                              void* d_out, int out_size, void* d_ws, size_t ws_size,
                              hipStream_t stream) {
    (void)in_sizes; (void)n_in; (void)out_size; (void)ws_size;
    const float* x       = (const float*)d_in[0];
    const float* c1a_w   = (const float*)d_in[1];
    const float* c1b_w   = (const float*)d_in[2];
    const float* bn1_g   = (const float*)d_in[3];
    const float* bn1_b   = (const float*)d_in[4];
    const float* c2_w    = (const float*)d_in[5];
    const float* bn2_g   = (const float*)d_in[6];
    const float* bn2_b   = (const float*)d_in[7];
    const float* c3_w    = (const float*)d_in[8];
    const float* bn3_g   = (const float*)d_in[9];
    const float* bn3_b   = (const float*)d_in[10];
    const float* s_ln_w  = (const float*)d_in[11];
    const float* s_conv_w= (const float*)d_in[12];
    const float* s_conv_b= (const float*)d_in[13];
    const float* s_Wi    = (const float*)d_in[14];
    const float* s_Wf    = (const float*)d_in[15];
    const float* s_Wz    = (const float*)d_in[16];
    const float* s_Wo    = (const float*)d_in[17];
    const float* s_R     = (const float*)d_in[18];
    const float* s_b     = (const float*)d_in[19];
    const float* s_gn_w  = (const float*)d_in[20];
    const float* f_ln_w  = (const float*)d_in[21];
    const float* f_up_w  = (const float*)d_in[22];
    const float* f_down_w= (const float*)d_in[23];
    const float* m_ln_w  = (const float*)d_in[24];
    const float* m_up_w  = (const float*)d_in[25];
    const float* m_conv_w= (const float*)d_in[26];
    const float* m_conv_b= (const float*)d_in[27];
    const float* m_Wq    = (const float*)d_in[28];
    const float* m_Wk    = (const float*)d_in[29];
    const float* m_Wv    = (const float*)d_in[30];
    const float* m_Wi    = (const float*)d_in[31];
    const float* m_bi    = (const float*)d_in[32];
    const float* m_Wf    = (const float*)d_in[33];
    const float* m_bf    = (const float*)d_in[34];
    const float* m_skip  = (const float*)d_in[35];
    const float* m_norm_w= (const float*)d_in[36];
    const float* m_down_w= (const float*)d_in[37];
    const float* post_ln_w=(const float*)d_in[38];
    const float* fc1_w   = (const float*)d_in[39];
    const float* fc1_b   = (const float*)d_in[40];
    const float* fc2_w   = (const float*)d_in[41];
    const float* fc2_b   = (const float*)d_in[42];
    float* out = (float*)d_out;

    // ---- workspace bump allocator ----
    char* ws = (char*)d_ws;
    size_t off = 0;
    auto alloc = [&](size_t bytes)->char* {
        char* p = ws + off;
        off += (bytes + 255) & ~(size_t)255;
        return p;
    };
    float* convA  = (float*)alloc((size_t)BBATCH*32*L1*sizeof(float));   // conv1/conv2/conv3 out, later ffmid
    float* convB  = (float*)alloc((size_t)BBATCH*32*P1*sizeof(float));   // pool1/pool2 out
    float* seq    = (float*)alloc((size_t)BS*DD*sizeof(float));
    float* stats  = (float*)alloc(256*sizeof(float));
    float* xibuf  = (float*)alloc((size_t)BS*DD*sizeof(float));
    float* xcv    = (float*)alloc((size_t)BS*DD*sizeof(float));
    float* gates  = (float*)alloc((size_t)BS*512*sizeof(float));         // also gup, up2
    float* hsbuf  = (float*)alloc((size_t)BS*DD*sizeof(float));
    unsigned short* bfA = (unsigned short*)alloc((size_t)BS*MI*sizeof(unsigned short));
    unsigned short* bfW = (unsigned short*)alloc((size_t)512*128*sizeof(unsigned short));
    float* xcv2   = (float*)alloc((size_t)BS*MI*sizeof(float));
    float* qbuf   = (float*)alloc((size_t)BS*MI*sizeof(float));          // later ht
    float* kbuf   = (float*)alloc((size_t)BS*MI*sizeof(float));
    float* vbuf   = (float*)alloc((size_t)BS*MI*sizeof(float));
    float* i_pre  = (float*)alloc((size_t)BBATCH*NHH*SL*sizeof(float));
    float* f_pre  = (float*)alloc((size_t)BBATCH*NHH*SL*sizeof(float));
    float* lfc    = (float*)alloc((size_t)BBATCH*NHH*SL*sizeof(float));
    float* mlD    = (float*)alloc((size_t)BBATCH*NHH*SL*sizeof(float));
    unsigned short* qhb = (unsigned short*)alloc((size_t)BBATCH*NHH*SPAD*MDH*sizeof(unsigned short));
    unsigned short* khb = (unsigned short*)alloc((size_t)BBATCH*NHH*SPAD*MDH*sizeof(unsigned short));
    float* vhp    = (float*)alloc((size_t)BBATCH*NHH*SPAD*MDH*sizeof(float));
    float* ffmid  = convA;   // conv buffers dead after seq built
    float* gup    = gates;
    float* up2    = gates;
    float* htbuf  = qbuf;

    // ---- conv front-end ----
    { long long t = (long long)BBATCH*32*L1;
      conv1_kernel<<<grid1d(t),256,0,stream>>>(x, c1a_w, c1b_w, convA, (int)t); }
    bn_stats_kernel<<<32,256,0,stream>>>(convA, stats, 32, L1);
    { long long t = (long long)BBATCH*32*P1;
      bn_pool_kernel<<<grid1d(t),256,0,stream>>>(convA, bn1_g, bn1_b, stats, convB, 32, L1, P1, 0, (int)t); }
    { long long t = (long long)BBATCH*64*L2A;
      conv3_kernel<<<grid1d(t),256,0,stream>>>(convB, c2_w, convA, 32, 64, P1, (int)t); }
    bn_stats_kernel<<<64,256,0,stream>>>(convA, stats, 64, L2A);
    { long long t = (long long)BBATCH*64*P2;
      bn_pool_kernel<<<grid1d(t),256,0,stream>>>(convA, bn2_g, bn2_b, stats, convB, 64, L2A, P2, 0, (int)t); }
    { long long t = (long long)BBATCH*128*L3A;
      conv3_kernel<<<grid1d(t),256,0,stream>>>(convB, c3_w, convA, 64, 128, P2, (int)t); }
    bn_stats_kernel<<<128,256,0,stream>>>(convA, stats, 128, L3A);
    { long long t = (long long)BBATCH*128*SL;
      bn_pool_kernel<<<grid1d(t),256,0,stream>>>(convA, bn3_g, bn3_b, stats, seq, 128, L3A, SL, 1, (int)t); }

    // ---- sLSTM block ----
    layernorm128<<<BS,128,0,stream>>>(seq, s_ln_w, xibuf, 1e-5f);
    { long long t = (long long)BS*DD;
      causal_dw_silu<<<grid1d(t),256,0,stream>>>(xibuf, DD, DD, s_conv_w, s_conv_b, xcv, (int)t); }
    { long long t = (long long)BS*512;
      slstm_gates_kernel<<<grid1d(t),256,0,stream>>>(xibuf, xcv, s_Wi, s_Wf, s_Wz, s_Wo, gates, (int)t); }
    slstm_scan_kernel<<<BBATCH*NHH,128,0,stream>>>(gates, s_R, s_b, hsbuf);
    slstm_headnorm_add<<<BS,128,0,stream>>>(hsbuf, s_gn_w, seq);

    // ---- FFN (GLU) via WMMA ----
    layernorm128<<<BS,128,0,stream>>>(seq, f_ln_w, xibuf, 1e-5f);
    f32_to_bf16_kernel<<<grid1d((long long)BS*128),256,0,stream>>>(xibuf, bfA, BS*128);
    transpose_to_bf16<<<grid1d(128*384),256,0,stream>>>(f_up_w, bfW, 128, 384);
    { int tiles = (BS/16)*(384/16);
      wmma_gemm_abT<<<(tiles+7)/8,256,0,stream>>>(bfA, bfW, gup, BS, 384, 128, 0); }
    ff_act_kernel<<<grid1d((long long)BS*FFD),256,0,stream>>>(gup, ffmid, BS*FFD);
    f32_to_bf16_kernel<<<grid1d((long long)BS*FFD),256,0,stream>>>(ffmid, bfA, BS*FFD);
    transpose_to_bf16<<<grid1d(192*128),256,0,stream>>>(f_down_w, bfW, 192, 128);
    { int tiles = (BS/16)*(128/16);
      wmma_gemm_abT<<<(tiles+7)/8,256,0,stream>>>(bfA, bfW, seq, BS, 128, 192, 1); }

    // ---- mLSTM block ----
    layernorm128<<<BS,128,0,stream>>>(seq, m_ln_w, xibuf, 1e-5f);
    f32_to_bf16_kernel<<<grid1d((long long)BS*128),256,0,stream>>>(xibuf, bfA, BS*128);
    transpose_to_bf16<<<grid1d(128*512),256,0,stream>>>(m_up_w, bfW, 128, 512);
    { int tiles = (BS/16)*(512/16);
      wmma_gemm_abT<<<(tiles+7)/8,256,0,stream>>>(bfA, bfW, up2, BS, 512, 128, 0); }
    { long long t = (long long)BS*MI;
      causal_dw_silu<<<grid1d(t),256,0,stream>>>(up2, 512, MI, m_conv_w, m_conv_b, xcv2, (int)t); }
    { long long t = (long long)BS*MI;
      bdiag_kernel<<<grid1d(t),256,0,stream>>>(xcv2, MI,  m_Wq, qbuf, (int)t);
      bdiag_kernel<<<grid1d(t),256,0,stream>>>(xcv2, MI,  m_Wk, kbuf, (int)t);
      bdiag_kernel<<<grid1d(t),256,0,stream>>>(up2,  512, m_Wv, vbuf, (int)t); }
    { long long t = (long long)BBATCH*NHH*SL;
      mlstm_ifpre<<<grid1d(t),256,0,stream>>>(qbuf, kbuf, vbuf, m_Wi, m_bi, m_Wf, m_bf, i_pre, f_pre, (int)t); }
    mlstm_scanstats<<<1,64,0,stream>>>(f_pre, i_pre, lfc, mlD);
    { long long t = (long long)BBATCH*NHH*SPAD*MDH;
      pack_heads_bf16<<<grid1d(t),256,0,stream>>>(qbuf, qhb, 0.125f, (int)t);   // fold 1/sqrt(64)
      pack_heads_bf16<<<grid1d(t),256,0,stream>>>(kbuf, khb, 1.0f,   (int)t);
      pack_heads_f32 <<<grid1d(t),256,0,stream>>>(vbuf, vhp, (int)t); }
    { size_t smem = (size_t)(16*SPAD + 16 + 8*MDH)*sizeof(float);   // 47168 B < 320 KB/WGP LDS
      mlstm_attn<<<BBATCH*NHH*STILES,256,smem,stream>>>(qhb, khb, vhp, i_pre, lfc, mlD, htbuf); }
    mlstm_post<<<BS,256,0,stream>>>(htbuf, m_norm_w, m_skip, xcv2, up2, bfA);
    transpose_to_bf16<<<grid1d(256*128),256,0,stream>>>(m_down_w, bfW, 256, 128);
    { int tiles = (BS/16)*(128/16);
      wmma_gemm_abT<<<(tiles+7)/8,256,0,stream>>>(bfA, bfW, seq, BS, 128, 256, 1); }

    // ---- post LN + classifier head ----
    layernorm128<<<BS,128,0,stream>>>(seq, post_ln_w, xibuf, 1e-5f);
    head_kernel<<<BBATCH,64,0,stream>>>(xibuf, fc1_w, fc1_b, fc2_w, fc2_b, out);
}
